// Model_9500467659086
// MI455X (gfx1250) — matrix operand order repaired and checked by reading, not compiled, not run on hardware
//
#include <hip/hip_runtime.h>
#include <hip/hip_bf16.h>

// Problem constants
#define BB 256
#define TT 240
#define TP 256   // T padded to multiple of 32 for WMMA K-loop
#define CC 512
#define KK 64
#define NCV 4

typedef __attribute__((ext_vector_type(16))) __bf16 v16bf;
typedef __attribute__((ext_vector_type(8)))  float  v8f;
typedef __attribute__((ext_vector_type(4)))  unsigned int v4u;
typedef __attribute__((ext_vector_type(8)))  int v8i;
typedef __attribute__((ext_vector_type(4)))  int v4i;

#if __has_builtin(__builtin_amdgcn_tensor_load_to_lds) && __has_builtin(__builtin_amdgcn_s_wait_tensorcnt)
#define HAS_TDM 1
#else
#define HAS_TDM 0
#endif

union Frag { v16bf v; unsigned u[8]; };

__device__ __forceinline__ unsigned short f2bf(float f) {
    union { float f; unsigned u; } x; x.f = f;
    unsigned u = x.u;
    u += 0x7FFFu + ((u >> 16) & 1u);   // round-to-nearest-even
    return (unsigned short)(u >> 16);
}

// ---------------------------------------------------------------------------
// Kernel 0: convert conv_w (f32 [64,512]) -> bf16
// ---------------------------------------------------------------------------
__global__ void kConv(const float* __restrict__ cw, unsigned short* __restrict__ cb) {
    int i = blockIdx.x * 256 + threadIdx.x;
    if (i < KK * CC) cb[i] = f2bf(cw[i]);
}

// ---------------------------------------------------------------------------
// Kernel A: normalize input rows, store xn (bf16, t-major, zero-padded to 256),
// compute logits via WMMA bf16, softmax -> a (f32, zero for padded rows).
// One wave (32 threads) handles a 16-row t-tile of one batch.
// ---------------------------------------------------------------------------
__global__ __launch_bounds__(32) void kA(const float* __restrict__ in,
                                         const unsigned short* __restrict__ convb,
                                         unsigned short* __restrict__ xn,
                                         float* __restrict__ a_out) {
    __shared__ unsigned short xt[16][CC];   // normalized bf16 tile [t-rows][c]
    __shared__ float lg[16][KK];            // logits tile

    const int lane = threadIdx.x;
    const int tile = blockIdx.x;            // 0..15
    const int b    = blockIdx.y;

    // ---- normalize: 2 lanes per row, 256 c each ----
    const int r = lane >> 1, h = lane & 1;
    const int t = tile * 16 + r;
    const bool valid = (t < TT);
    const float* row = in + ((size_t)b * TT + (valid ? t : 0)) * CC + h * 256;

    float ss = 0.f;
    if (valid) {
        for (int c = 0; c < 256; c += 4) {
            float4 v = *(const float4*)(row + c);
            ss += v.x*v.x + v.y*v.y + v.z*v.z + v.w*v.w;
        }
    }
    ss += __shfl_xor(ss, 1);
    const float inv = 1.f / fmaxf(sqrtf(ss), 1e-12f);

    unsigned short* xrow = xn + ((size_t)b * TP + t) * CC + h * 256;
    for (int c = 0; c < 256; c += 4) {
        float4 v = valid ? *(const float4*)(row + c) : make_float4(0.f, 0.f, 0.f, 0.f);
        unsigned p0 = (unsigned)f2bf(v.x * inv) | ((unsigned)f2bf(v.y * inv) << 16);
        unsigned p1 = (unsigned)f2bf(v.z * inv) | ((unsigned)f2bf(v.w * inv) << 16);
        *(unsigned*)&xt[r][h * 256 + c]     = p0;
        *(unsigned*)&xt[r][h * 256 + c + 2] = p1;
        *(unsigned*)(xrow + c)     = p0;
        *(unsigned*)(xrow + c + 2) = p1;
    }
    __syncthreads();

    // ---- logits = xn_tile [16 x 512] * conv_w^T [512 x 64] via WMMA bf16 ----
    const int m = lane & 15, half = lane >> 4;
    for (int nt = 0; nt < 4; ++nt) {
        v8f acc = {0.f, 0.f, 0.f, 0.f, 0.f, 0.f, 0.f, 0.f};
        for (int kb = 0; kb < 16; ++kb) {
            const int cb = kb * 32;
            Frag af, bf_;
#pragma unroll
            for (int v = 0; v < 8; ++v) {
                const int ka = cb + ((v >= 4) ? 16 : 0) + half * 8 + (v & 3) * 2;
                af.u[v] = *(const unsigned*)&xt[m][ka];
                const int kc = cb + ((v >= 4) ? 16 : 0) + half * 8 + (v & 3) * 2;      // reduction index (c)
                bf_.u[v] = *(const unsigned*)&convb[(size_t)(nt * 16 + m) * CC + kc];
            }
            acc = __builtin_amdgcn_wmma_f32_16x16x32_bf16(
                false, af.v, false, bf_.v, (short)0, acc, false, false);
        }
#pragma unroll
        for (int j = 0; j < 8; ++j)
            lg[j + half * 8][nt * 16 + m] = acc[j];
    }
    __syncthreads();

    // ---- softmax over 64 clusters: 2 lanes per row, 32 values each ----
    const float* lr = &lg[r][h * 32];
    float mx = -1e30f;
#pragma unroll
    for (int i = 0; i < 32; ++i) mx = fmaxf(mx, lr[i]);
    mx = fmaxf(mx, __shfl_xor(mx, 1));
    float e[32];
    float s = 0.f;
#pragma unroll
    for (int i = 0; i < 32; ++i) { e[i] = __expf(lr[i] - mx); s += e[i]; }
    s += __shfl_xor(s, 1);
    const float invs = valid ? (1.f / s) : 0.f;   // padded rows get a == 0
    float* arow = a_out + ((size_t)b * TP + t) * KK + h * 32;
#pragma unroll
    for (int i = 0; i < 32; ++i) arow[i] = e[i] * invs;
}

// ---------------------------------------------------------------------------
// Kernel B: per batch b: vlad = a^T x - asum*centroids, intra-norm, final norm.
// 512 threads = 16 waves; wave -> (ktile 0..3, cchunk 0..3 of 128 columns).
// The per-iteration `a` tile (32x64 f32, 8KB) is DMA'd into LDS by the
// Tensor Data Mover when available (tensor_load_to_lds + s_wait_tensorcnt).
// ---------------------------------------------------------------------------
__global__ __launch_bounds__(512) void kB(const unsigned short* __restrict__ xn,
                                          const float* __restrict__ a_in,
                                          const float* __restrict__ centroids,
                                          float* __restrict__ v_out) {
    __shared__ unsigned short aT[KK][32];   // a transposed tile, bf16
    __shared__ unsigned short xT[CC][32];   // xn transposed tile, bf16
    __shared__ float asp[8][KK];
    __shared__ float asum[KK];
    __shared__ float rown[KK];
    __shared__ float vnorm;
#if HAS_TDM
    __shared__ float astage[32 * KK];       // TDM landing buffer, natural [t][k]
#endif

    const int tid  = threadIdx.x;
    const int b    = blockIdx.x;
    const int lane = tid & 31;
    const int w    = tid >> 5;
    const int ktile  = w >> 2;
    const int cchunk = w & 3;
    const int m = lane & 15, half = lane >> 4;

    const float* ab = a_in + (size_t)b * TP * KK;
    const unsigned short* xb = xn + (size_t)b * TP * CC;

    // asum[k] = sum_t a[b,t,k]
    {
        const int k = tid & 63, part = tid >> 6;
        float s = 0.f;
        for (int t2 = 0; t2 < 32; ++t2) s += ab[(size_t)(part * 32 + t2) * KK + k];
        asp[part][k] = s;
    }
    __syncthreads();
    if (tid < KK) {
        float s = 0.f;
#pragma unroll
        for (int p = 0; p < 8; ++p) s += asp[p][tid];
        asum[tid] = s;
        rown[tid] = 0.f;
    }
    __syncthreads();

    v8f acc[8];
#pragma unroll
    for (int nt = 0; nt < 8; ++nt) acc[nt] = (v8f){0.f,0.f,0.f,0.f,0.f,0.f,0.f,0.f};

    for (int kb = 0; kb < 8; ++kb) {
        const int tb = kb * 32;

#if HAS_TDM
        // --- Tensor Data Mover: DMA a[tb..tb+31][0..63] f32 -> LDS astage ---
        if (w == 0) {
            const unsigned lds_off = (unsigned)(uintptr_t)(void*)astage;
            const unsigned long long ga =
                (unsigned long long)(uintptr_t)(ab + (size_t)tb * KK);
            v4u g0;
            g0[0] = 1u;                        // count=1, user descriptor
            g0[1] = lds_off;                   // lds_addr
            g0[2] = (unsigned)ga;              // global_addr[31:0]
            g0[3] = (unsigned)((ga >> 32) & 0x1FFFFFFu) | (2u << 30); // addr[56:32], type=2
            v8i g1;
            g1[0] = (2 << 16);                 // data_size = 4 bytes
            g1[1] = (int)((KK & 0xFFFF) << 16);// tensor_dim0 = 64 (bits 63:48)
            g1[2] = (32 << 16);                // tensor_dim1 = 32 (bits 95:80)
            g1[3] = (KK << 16);                // tile_dim0 = 64 (bits 127:112)
            g1[4] = 32;                        // tile_dim1 = 32 (bits 143:128)
            g1[5] = KK;                        // tensor_dim0_stride = 64 (bits 191:160)
            g1[6] = 0;
            g1[7] = 0;
            v4i z4 = {0, 0, 0, 0};
#if defined(__clang_major__) && (__clang_major__ >= 23)
            v8i z8 = {0, 0, 0, 0, 0, 0, 0, 0};
            __builtin_amdgcn_tensor_load_to_lds(g0, g1, z4, z4, z8, 0);
#else
            __builtin_amdgcn_tensor_load_to_lds(g0, g1, z4, z4, 0);
#endif
            __builtin_amdgcn_s_wait_tensorcnt(0);
        }
        __syncthreads();   // publish astage to all waves
#endif

        // stage a^T (bf16): 2048 elems, 4 per thread
        {
            const int i0 = tid * 4;
#pragma unroll
            for (int q = 0; q < 4; ++q) {
                const int idx = i0 + q;
                const int t2 = idx >> 6, k = idx & 63;
#if HAS_TDM
                aT[k][t2] = f2bf(astage[idx]);   // astage is [t2][k] == linear idx
#else
                aT[k][t2] = f2bf(ab[(size_t)(tb + t2) * KK + k]);
#endif
            }
        }
        // stage xn^T (bf16): 32x512 elems, 32 per thread
        {
            const int t2 = tid >> 4, cg = tid & 15;
            const unsigned short* xr = xb + (size_t)(tb + t2) * CC + cg * 32;
            __builtin_prefetch(xr + CC * 32, 0, 0);   // next kb iteration's rows
#pragma unroll
            for (int j = 0; j < 32; ++j) xT[cg * 32 + j][t2] = xr[j];
        }
        __syncthreads();

        Frag af;
#pragma unroll
        for (int v = 0; v < 8; ++v) {
            const int ti = ((v >= 4) ? 16 : 0) + half * 8 + (v & 3) * 2;
            af.u[v] = *(const unsigned*)&aT[ktile * 16 + m][ti];
        }
#pragma unroll
        for (int nt = 0; nt < 8; ++nt) {
            Frag bf_;
            const int c = cchunk * 128 + nt * 16 + m;
#pragma unroll
            for (int v = 0; v < 8; ++v) {
                const int ti = ((v >= 4) ? 16 : 0) + half * 8 + (v & 3) * 2;
                bf_.u[v] = *(const unsigned*)&xT[c][ti];
            }
            acc[nt] = __builtin_amdgcn_wmma_f32_16x16x32_bf16(
                false, af.v, false, bf_.v, (short)0, acc[nt], false, false);
        }
        __syncthreads();
    }

    // subtract asum*centroid, accumulate per-row squared norms
#pragma unroll
    for (int nt = 0; nt < 8; ++nt) {
        const int c = cchunk * 128 + nt * 16 + m;
#pragma unroll
        for (int j = 0; j < 8; ++j) {
            const int k = ktile * 16 + j + half * 8;
            float val = acc[nt][j] - asum[k] * centroids[(size_t)k * CC + c];
            acc[nt][j] = val;
            float sq = val * val;
            sq += __shfl_xor(sq, 1);
            sq += __shfl_xor(sq, 2);
            sq += __shfl_xor(sq, 4);
            sq += __shfl_xor(sq, 8);
            if (m == 0) atomicAdd(&rown[k], sq);
        }
    }
    __syncthreads();
    if (tid < KK) {
        const float rr = rown[tid];
        const float ir = 1.f / fmaxf(sqrtf(rr), 1e-12f);
        rown[tid] = ir;                 // now holds inverse row norm
        asp[0][tid] = rr * ir * ir;     // contribution to final vector norm
    }
    __syncthreads();
    if (tid == 0) {
        float s = 0.f;
        for (int k2 = 0; k2 < KK; ++k2) s += asp[0][k2];
        vnorm = 1.f / fmaxf(sqrtf(s), 1e-12f);
    }
    __syncthreads();
    const float fin = vnorm;

    float* vb = v_out + (size_t)b * (KK * CC);
#pragma unroll
    for (int nt = 0; nt < 8; ++nt) {
        const int c = cchunk * 128 + nt * 16 + m;
#pragma unroll
        for (int j = 0; j < 8; ++j) {
            const int k = ktile * 16 + j + half * 8;
            vb[(size_t)k * CC + c] = acc[nt][j] * rown[k] * fin;
        }
    }
}

// ---------------------------------------------------------------------------
// Kernel C: out[b,n] = sigmoid(v[b,:] . fc_w[n,:] + fc_b[n])
// ---------------------------------------------------------------------------
__global__ __launch_bounds__(256) void kC(const float* __restrict__ v,
                                          const float* __restrict__ fcw,
                                          const float* __restrict__ fcb,
                                          float* __restrict__ out) {
    __shared__ float red[256];
    const int b = blockIdx.x, tid = threadIdx.x;
    const float* vb = v + (size_t)b * (KK * CC);
    float acc[NCV] = {0.f, 0.f, 0.f, 0.f};
    for (int j = tid; j < KK * CC; j += 256) {
        __builtin_prefetch(vb + j + 1024, 0, 0);
        __builtin_prefetch(fcw + j + 1024, 0, 0);
        const float x = vb[j];
#pragma unroll
        for (int n = 0; n < NCV; ++n) acc[n] += x * fcw[(size_t)n * (KK * CC) + j];
    }
    for (int n = 0; n < NCV; ++n) {
        red[tid] = acc[n];
        __syncthreads();
        for (int s = 128; s > 0; s >>= 1) {
            if (tid < s) red[tid] += red[tid + s];
            __syncthreads();
        }
        if (tid == 0) {
            const float z = red[0] + fcb[n];
            out[b * NCV + n] = 1.f / (1.f + __expf(-z));
        }
        __syncthreads();
    }
}

// ---------------------------------------------------------------------------
extern "C" void kernel_launch(void* const* d_in, const int* in_sizes, int n_in,
                              void* d_out, int out_size, void* d_ws, size_t ws_size,
                              hipStream_t stream) {
    const float* inputs    = (const float*)d_in[0];   // [B,T,C]
    const float* conv_w    = (const float*)d_in[1];   // [K,C]
    const float* centroids = (const float*)d_in[2];   // [K,C]
    const float* fc_w      = (const float*)d_in[3];   // [NC,K*C]
    const float* fc_b      = (const float*)d_in[4];   // [NC]
    float* out = (float*)d_out;                       // [B,NC]

    char* ws = (char*)d_ws;
    size_t off = 0;
    unsigned short* convb = (unsigned short*)(ws + off); off += (size_t)KK * CC * 2;        // 64 KB
    unsigned short* xn    = (unsigned short*)(ws + off); off += (size_t)BB * TP * CC * 2;   // 64 MB
    float*          a_buf = (float*)(ws + off);          off += (size_t)BB * TP * KK * 4;   // 16 MB
    float*          v_buf = (float*)(ws + off);          off += (size_t)BB * KK * CC * 4;   // 32 MB
    (void)off; (void)ws_size; (void)in_sizes; (void)n_in; (void)out_size;

    kConv<<<(KK * CC + 255) / 256, 256, 0, stream>>>(conv_w, convb);
    kA<<<dim3(16, BB), 32, 0, stream>>>(inputs, convb, xn, a_buf);
    kB<<<BB, 512, 0, stream>>>(xn, a_buf, centroids, v_buf);
    kC<<<BB, 256, 0, stream>>>(v_buf, fc_w, fc_b, out);
}
